// GraphGPSNet_85761906967229
// MI455X (gfx1250) — matrix operand order repaired and checked by reading, not compiled, not run on hardware
//
#include <hip/hip_runtime.h>
#include <hip/hip_bf16.h>

// GraphGPS on gfx1250 (MI455X):
//  - all GEMMs via v_wmma_f32_16x16x32_f16 (fp32 accumulate)
//  - weights pre-swizzled once into WMMA B-fragment layout (f16, contiguous
//    32B per lane -> 2x global_load_b128, zero packing ALU)
//  - A tiles staged with global_load_async_to_lds_b128 + s_wait_asynccnt
//  - fused 2-layer MLPs staged through per-wave LDS f16 tiles
//  - flash attention (one wave per 16-query tile); V kept transposed so P*V
//    B-fragments are contiguous too
//  - GINE edge GEMM fused with relu + global_atomic_add_f32 scatter

#define DEVINL __device__ __forceinline__

typedef __attribute__((ext_vector_type(16))) _Float16 v16h;
typedef __attribute__((ext_vector_type(8)))  float    v8f;

namespace gps {

constexpr int kN = 32768, kS = 1024, kB = 32, kE = 262144, kC = 128, kL = 4;
constexpr float kLnEps = 1e-5f;

enum { ACT_NONE = 0, ACT_RELU = 1, ACT_GELU = 2 };

DEVINL v8f wmma16x16x32(v16h a, v16h b, v8f c) {
  // (neg_a, A, neg_b, B, c_mod, C, reuse_a, reuse_b)
  return __builtin_amdgcn_wmma_f32_16x16x32_f16(false, a, false, b, (short)0, c,
                                                false, false);
}

DEVINL v8f zero8() {
  v8f z;
#pragma unroll
  for (int r = 0; r < 8; ++r) z[r] = 0.0f;
  return z;
}

DEVINL void wait_ds() { asm volatile("s_wait_dscnt 0x0" ::: "memory"); }
DEVINL void wait_async() { asm volatile("s_wait_asynccnt 0x0" ::: "memory"); }

// Generic (flat) pointer to a __shared__ object: LDS offset lives in the low
// 32 bits of the shared-aperture address (ISA §10.2 LDS addressing).
DEVINL uint32_t lds_off32(const void* p) { return (uint32_t)(size_t)p; }

// ---- async staging of one 16x32 A tile into per-wave LDS -------------------
// f32 tile: 2 KB -> 4 x b128 per lane. f16 tile: 1 KB -> 2 x b128 per lane.
DEVINL void stage_a32_async(const float* __restrict__ A, int lda, int row0,
                            int k0, float* __restrict__ tile) {
  const int lane = threadIdx.x & 31;
  const int elem = lane << 4;                 // 16 floats per lane
  const int row = elem >> 5, col = elem & 31; // row-major 16x32 tile
  const float* gp = A + (size_t)(row0 + row) * lda + (k0 + col);
  const uint32_t la = lds_off32(tile + elem);
#pragma unroll
  for (int c = 0; c < 4; ++c)
    asm volatile("global_load_async_to_lds_b128 %0, %1, off"
                 :: "v"(la + 16u * c), "v"(gp + 4 * c) : "memory");
}

DEVINL void stage_a16_async(const _Float16* __restrict__ A, int lda, int row0,
                            int k0, _Float16* __restrict__ tile) {
  const int lane = threadIdx.x & 31;
  const int elem = lane << 4;                 // 16 halfs per lane
  const int row = elem >> 5, col = elem & 31;
  const _Float16* gp = A + (size_t)(row0 + row) * lda + (k0 + col);
  const uint32_t la = lds_off32(tile + elem);
#pragma unroll
  for (int c = 0; c < 2; ++c)
    asm volatile("global_load_async_to_lds_b128 %0, %1, off"
                 :: "v"(la + 16u * c), "v"(gp + 8 * c) : "memory");
}

// --- WMMA fragment loaders (layouts per CDNA5 ISA §7.12.2) ------
// A fragment from a row-major 16x32 tile (f32 in LDS).
DEVINL v16h frag_a_tile32(const float* __restrict__ tile) {
  const int lane = threadIdx.x & 31;
  const float* ap = tile + (lane & 15) * 32;
  const int kb = (lane >> 4) << 3;
  v16h a;
#pragma unroll
  for (int e = 0; e < 16; ++e) a[e] = (_Float16)ap[kb + ((e < 8) ? e : (e + 8))];
  return a;
}
// A fragment from a row-major 16x32 tile (f16 in LDS).
DEVINL v16h frag_a_tile16(const _Float16* __restrict__ tile) {
  const int lane = threadIdx.x & 31;
  const _Float16* ap = tile + (lane & 15) * 32;
  const int kb = (lane >> 4) << 3;
  v16h a;
#pragma unroll
  for (int e = 0; e < 16; ++e) a[e] = ap[kb + ((e < 8) ? e : (e + 8))];
  return a;
}
// A fragment from an f16 row-major matrix with arbitrary row stride.
DEVINL v16h frag_a_lds(const _Float16* __restrict__ A, int lda, int k0) {
  const int lane = threadIdx.x & 31;
  const _Float16* ap = A + (lane & 15) * lda;
  const int kb = k0 + ((lane >> 4) << 3);
  v16h a;
#pragma unroll
  for (int e = 0; e < 16; ++e) a[e] = ap[kb + ((e < 8) ? e : (e + 8))];
  return a;
}
// A fragment straight from global f32 with K bounds mask (partial tiles).
DEVINL v16h frag_a_f32(const float* __restrict__ A, int lda, int row0, int k0,
                       int Kreal) {
  const int lane = threadIdx.x & 31;
  const float* ap = A + (size_t)(row0 + (lane & 15)) * lda;
  const int kb = k0 + ((lane >> 4) << 3);
  v16h a;
#pragma unroll
  for (int e = 0; e < 16; ++e) {
    const int k = kb + ((e < 8) ? e : (e + 8));
    a[e] = (k < Kreal) ? (_Float16)ap[k] : (_Float16)0.0f;
  }
  return a;
}
// A fragment from global f16 (attention Q).
DEVINL v16h frag_a_f16(const _Float16* __restrict__ A, int lda, int row0,
                       int k0) {
  const int lane = threadIdx.x & 31;
  const _Float16* ap = A + (size_t)(row0 + (lane & 15)) * lda;
  const int kb = k0 + ((lane >> 4) << 3);
  v16h a;
#pragma unroll
  for (int e = 0; e < 16; ++e) a[e] = ap[kb + ((e < 8) ? e : (e + 8))];
  return a;
}
// B fragment from the swizzled weight arena: layout [kt][nt][lane][16 halfs],
// i.e. each lane reads 32 contiguous bytes (2 x b128), coalesced per wave.
DEVINL v16h frag_b_swz(const _Float16* __restrict__ Ws, int ntn, int kt,
                       int nt) {
  const int lane = threadIdx.x & 31;
  const _Float16* p = Ws + (((size_t)kt * ntn + nt) * 32 + lane) * 16;
  v16h b;
#pragma unroll
  for (int e = 0; e < 16; ++e) b[e] = p[e];
  return b;
}
// B fragment loaded transposed from row-major f16: B[k][n] = M[n0+n][k].
// Used for Q*K^T (keys row-contiguous) and P*V with the transposed V stream.
DEVINL v16h frag_bT_f16(const _Float16* __restrict__ M, int ld, int n0,
                        int k0) {
  const int lane = threadIdx.x & 31;
  const _Float16* p =
      M + (size_t)(n0 + (lane & 15)) * ld + k0 + ((lane >> 4) << 3);
  v16h b;
#pragma unroll
  for (int e = 0; e < 16; ++e) b[e] = p[(e < 8) ? e : (e + 8)];
  return b;
}

DEVINL float gelu_tanh(float v) {
  return 0.5f * v * (1.0f + tanhf(0.7978845608028654f * (v + 0.044715f * v * v * v)));
}
template <int ACT> DEVINL float apply_act(float v) {
  if (ACT == ACT_RELU) return fmaxf(v, 0.0f);
  if (ACT == ACT_GELU) return gelu_tanh(v);
  return v;
}

// ---------------------------------------------------------------------------
// Fused 2-layer MLP: out = act(A[M,K1] @ W1[K1,HID] + b1) @ W2[HID,128] + b2
//                    (+ res). Hidden tile lives only in per-wave LDS (f16).
// Weights are swizzled (K zero-padded to x32). Block = 128 = 4 waves.
// ---------------------------------------------------------------------------
template <int HID, int ACT, bool OUTF16>
__global__ void __launch_bounds__(128)
mlp2_kernel(const float* __restrict__ A, int lda, int K1,
            const _Float16* __restrict__ W1s, const float* __restrict__ B1,
            const _Float16* __restrict__ W2s, const float* __restrict__ B2,
            const float* __restrict__ res, void* __restrict__ out, int M) {
  constexpr int NT = HID / 16;
  __shared__ _Float16 hbuf[4][16 * HID];
  __shared__ float atile[4][16 * 32];
  const int wv = threadIdx.x >> 5, lane = threadIdx.x & 31;
  const int row0 = (blockIdx.x * 4 + wv) * 16;
  if (row0 >= M) return;
  const int n = lane & 15, mb = (lane >> 4) << 3;
  float* at = &atile[wv][0];
  const int KT1 = (K1 + 31) / 32;

  // ---- layer 1 ----
  v8f acc1[NT];
#pragma unroll
  for (int t = 0; t < NT; ++t) {
    const float bv = B1[t * 16 + n];
#pragma unroll
    for (int r = 0; r < 8; ++r) acc1[t][r] = bv;
  }
  for (int kt = 0; kt < KT1; ++kt) {
    const int kk = kt * 32;
    v16h a;
    if (kk + 32 <= K1) {                  // full tile: async-stage via LDS
      wait_ds();
      stage_a32_async(A, lda, row0, kk, at);
      wait_async();
      a = frag_a_tile32(at);
    } else {                              // ragged K tail: masked scalar path
      a = frag_a_f32(A, lda, row0, kk, K1);
    }
#pragma unroll
    for (int t = 0; t < NT; ++t)
      acc1[t] = wmma16x16x32(a, frag_b_swz(W1s, NT, kt, t), acc1[t]);
  }
  // act + scatter C-layout -> row-major f16 LDS tile (16 x HID)
  _Float16* hb = &hbuf[wv][0];
#pragma unroll
  for (int t = 0; t < NT; ++t)
#pragma unroll
    for (int r = 0; r < 8; ++r)
      hb[(mb + r) * HID + t * 16 + n] = (_Float16)apply_act<ACT>(acc1[t][r]);
  wait_ds();  // cross-lane LDS RAW: wave stores visible before A-frag loads

  // ---- layer 2 ----
  v8f acc2[8];
#pragma unroll
  for (int t = 0; t < 8; ++t) {
    const float bv = B2[t * 16 + n];
#pragma unroll
    for (int r = 0; r < 8; ++r) acc2[t][r] = bv;
  }
#pragma unroll
  for (int kt = 0; kt < HID / 32; ++kt) {
    if (kt + 1 < HID / 32)
      __builtin_prefetch(W2s + ((size_t)(kt + 1) * 8 * 32 + lane) * 16, 0, 1);
    const v16h a = frag_a_lds(hb, HID, kt * 32);  // hidden tile, stride HID
#pragma unroll
    for (int t = 0; t < 8; ++t)
      acc2[t] = wmma16x16x32(a, frag_b_swz(W2s, 8, kt, t), acc2[t]);
  }
#pragma unroll
  for (int t = 0; t < 8; ++t)
#pragma unroll
    for (int r = 0; r < 8; ++r) {
      const size_t idx = (size_t)(row0 + mb + r) * 128 + t * 16 + n;
      float v = acc2[t][r];
      if (res) v += res[idx];
      if (OUTF16) ((_Float16*)out)[idx] = (_Float16)v;
      else ((float*)out)[idx] = v;
    }
}

// ---------------------------------------------------------------------------
// GEMM over a 128-column window: out[:, cb..cb+128) = A[M,128] @ W[:, cb..]
// + bias. Swizzled weights (ntn = total columns / 16). Block = 128 = 4 waves.
// ---------------------------------------------------------------------------
template <int ACT, bool OUTF16>
__global__ void __launch_bounds__(128)
gemm128_kernel(const float* __restrict__ A, int lda,
               const _Float16* __restrict__ Ws, int ntn,
               const float* __restrict__ bias, const float* __restrict__ res,
               void* __restrict__ out, int ldo, int M) {
  __shared__ float atile[4][16 * 32];
  const int wv = threadIdx.x >> 5, lane = threadIdx.x & 31;
  const int row0 = (blockIdx.x * 4 + wv) * 16;
  if (row0 >= M) return;
  const int cb = blockIdx.y * 128;
  const int n = lane & 15, mb = (lane >> 4) << 3;
  float* at = &atile[wv][0];
  v8f acc[8];
#pragma unroll
  for (int t = 0; t < 8; ++t) {
    const float bv = bias[cb + t * 16 + n];
#pragma unroll
    for (int r = 0; r < 8; ++r) acc[t][r] = bv;
  }
#pragma unroll
  for (int kt = 0; kt < 4; ++kt) {
    wait_ds();
    stage_a32_async(A, lda, row0, kt * 32, at);
    if (kt < 3)
      __builtin_prefetch(
          Ws + (((size_t)(kt + 1) * ntn + blockIdx.y * 8) * 32 + lane) * 16, 0, 1);
    wait_async();
    const v16h a = frag_a_tile32(at);
#pragma unroll
    for (int t = 0; t < 8; ++t)
      acc[t] = wmma16x16x32(a, frag_b_swz(Ws, ntn, kt, blockIdx.y * 8 + t),
                            acc[t]);
  }
#pragma unroll
  for (int t = 0; t < 8; ++t)
#pragma unroll
    for (int r = 0; r < 8; ++r) {
      const size_t idx = (size_t)(row0 + mb + r) * ldo + cb + t * 16 + n;
      float v = apply_act<ACT>(acc[t][r]);
      if (res) v += res[idx];
      if (OUTF16) ((_Float16*)out)[idx] = (_Float16)v;
      else ((float*)out)[idx] = v;
    }
}

// ---------------------------------------------------------------------------
// GINE fused: m = relu(x[src] + e @ elin_w + b); agg[dst] += m (f32 atomics).
// e is f16 [E,128]; wave owns 16 edges x all 128 cols; A async-staged.
// ---------------------------------------------------------------------------
__global__ void __launch_bounds__(128)
gine_kernel(const _Float16* __restrict__ e, const _Float16* __restrict__ Ws,
            const float* __restrict__ bias, const float* __restrict__ x,
            const int* __restrict__ src, const int* __restrict__ dst,
            float* __restrict__ agg) {
  __shared__ _Float16 atile[4][16 * 32];
  const int wv = threadIdx.x >> 5, lane = threadIdx.x & 31;
  const int row0 = (blockIdx.x * 4 + wv) * 16;
  const int n = lane & 15, mb = (lane >> 4) << 3;
  _Float16* at = &atile[wv][0];
  v8f acc[8];
#pragma unroll
  for (int t = 0; t < 8; ++t) {
    const float bv = bias[t * 16 + n];
#pragma unroll
    for (int r = 0; r < 8; ++r) acc[t][r] = bv;
  }
#pragma unroll
  for (int kt = 0; kt < 4; ++kt) {
    wait_ds();
    stage_a16_async(e, 128, row0, kt * 32, at);
    wait_async();
    const v16h a = frag_a_tile16(at);
#pragma unroll
    for (int t = 0; t < 8; ++t)
      acc[t] = wmma16x16x32(a, frag_b_swz(Ws, 8, kt, t), acc[t]);
  }
#pragma unroll
  for (int r = 0; r < 8; ++r) {
    const int eg = row0 + mb + r;
    const int s = src[eg], d = dst[eg];
#pragma unroll
    for (int t = 0; t < 8; ++t) {
      const int c = t * 16 + n;
      const float v = fmaxf(acc[t][r] + x[(size_t)s * 128 + c], 0.0f);
      __hip_atomic_fetch_add(&agg[(size_t)d * 128 + c], v, __ATOMIC_RELAXED,
                             __HIP_MEMORY_SCOPE_AGENT);
    }
  }
}

// ---------------------------------------------------------------------------
// Flash attention: one wave per (graph, head, 16-query tile), f16 Q/K + Vt.
// DH=32 is one A fragment; 32 keys per step (2 score + 2 PV WMMAs).
// ---------------------------------------------------------------------------
__global__ void __launch_bounds__(256)
attn_kernel(const _Float16* __restrict__ qkv, const _Float16* __restrict__ vt,
            float* __restrict__ o) {
  __shared__ _Float16 plds[8][16 * 32];
  const int wv = threadIdx.x >> 5, lane = threadIdx.x & 31;
  const int w = blockIdx.x * 8 + wv;
  const int qt = w & 63, h = (w >> 6) & 3, b = w >> 8;
  const _Float16* base = qkv + (size_t)b * kS * 384;
  const _Float16* Qp = base + h * 32;
  const _Float16* Kp = base + 128 + h * 32;
  const _Float16* Vt = vt + ((size_t)b * 128 + h * 32) * kS;  // [d][s]
  const int n = lane & 15, mb = (lane >> 4) << 3;
  const float scale = 0.17677669529663687f;  // 1/sqrt(32)

  const v16h qf = frag_a_f16(Qp, 384, qt * 16, 0);
  float rmax[8], rsum[8];
  v8f o0 = zero8(), o1 = zero8();
#pragma unroll
  for (int r = 0; r < 8; ++r) { rmax[r] = -1e30f; rsum[r] = 0.0f; }
  _Float16* pb = &plds[wv][0];

  for (int kb = 0; kb < kS; kb += 32) {
    v8f s0 = wmma16x16x32(qf, frag_bT_f16(Kp, 384, kb, 0), zero8());
    v8f s1 = wmma16x16x32(qf, frag_bT_f16(Kp, 384, kb + 16, 0), zero8());
#pragma unroll
    for (int r = 0; r < 8; ++r) { s0[r] *= scale; s1[r] *= scale; }
    float tm[8];
#pragma unroll
    for (int r = 0; r < 8; ++r) tm[r] = fmaxf(s0[r], s1[r]);
#pragma unroll
    for (int off = 1; off < 16; off <<= 1)
#pragma unroll
      for (int r = 0; r < 8; ++r) tm[r] = fmaxf(tm[r], __shfl_xor(tm[r], off, 32));
#pragma unroll
    for (int r = 0; r < 8; ++r) {
      const float nm = fmaxf(rmax[r], tm[r]);
      const float al = __expf(rmax[r] - nm);
      rmax[r] = nm;
      const float p0 = __expf(s0[r] - nm);
      const float p1 = __expf(s1[r] - nm);
      float ts = p0 + p1;
#pragma unroll
      for (int off = 1; off < 16; off <<= 1) ts += __shfl_xor(ts, off, 32);
      rsum[r] = rsum[r] * al + ts;
      o0[r] *= al;
      o1[r] *= al;
      pb[(mb + r) * 32 + n] = (_Float16)p0;
      pb[(mb + r) * 32 + 16 + n] = (_Float16)p1;
    }
    wait_ds();  // cross-lane LDS RAW before building the P fragment
    const v16h pf = frag_a_tile16(pb);
    // Vt fragment: B[k][n] = V[kb+k][d=n] = Vt[d][kb+k] -> contiguous per lane
    o0 = wmma16x16x32(pf, frag_bT_f16(Vt, kS, 0, kb), o0);
    o1 = wmma16x16x32(pf, frag_bT_f16(Vt, kS, 16, kb), o1);
  }
#pragma unroll
  for (int r = 0; r < 8; ++r) {
    const float inv = 1.0f / rsum[r];
    const size_t node = (size_t)b * kS + qt * 16 + mb + r;
    o[node * 128 + h * 32 + n] = o0[r] * inv;
    o[node * 128 + h * 32 + 16 + n] = o1[r] * inv;
  }
}

// --------------------------- elementwise / LN ------------------------------
__global__ void ln_kernel(const float* __restrict__ a,
                          const float* __restrict__ res,
                          const float* __restrict__ g,
                          const float* __restrict__ bta,
                          float* __restrict__ out, int accum, int M) {
  const int wv = threadIdx.x >> 5, lane = threadIdx.x & 31;
  const int row = blockIdx.x * 8 + wv;
  if (row >= M) return;
  const size_t off = (size_t)row * 128;
  float v[4], s = 0.0f, sq = 0.0f;
#pragma unroll
  for (int i = 0; i < 4; ++i) {
    const int c = lane + 32 * i;
    v[i] = a[off + c];
    if (res) v[i] += res[off + c];
    s += v[i];
    sq += v[i] * v[i];
  }
#pragma unroll
  for (int o2 = 1; o2 < 32; o2 <<= 1) {
    s += __shfl_xor(s, o2, 32);
    sq += __shfl_xor(sq, o2, 32);
  }
  const float mu = s * (1.0f / 128.0f);
  const float var = sq * (1.0f / 128.0f) - mu * mu;
  const float rstd = rsqrtf(var + kLnEps);
#pragma unroll
  for (int i = 0; i < 4; ++i) {
    const int c = lane + 32 * i;
    const float y = (v[i] - mu) * rstd * g[c] + bta[c];
    if (accum) out[off + c] += y; else out[off + c] = y;
  }
}

__global__ void zero_kernel(float* __restrict__ p, size_t nel) {
  const size_t i = (size_t)blockIdx.x * blockDim.x + threadIdx.x;
  if (i < nel) p[i] = 0.0f;
}
__global__ void add_kernel(const float* __restrict__ a,
                           const float* __restrict__ b,
                           float* __restrict__ out, size_t nel) {
  const size_t i = (size_t)blockIdx.x * blockDim.x + threadIdx.x;
  if (i < nel) out[i] = a[i] + b[i];
}
__global__ void ginpre_kernel(const float* __restrict__ x,
                              const float* __restrict__ agg,
                              const float* __restrict__ eps, int layer,
                              float* __restrict__ out, size_t nel) {
  const size_t i = (size_t)blockIdx.x * blockDim.x + threadIdx.x;
  if (i < nel) out[i] = (1.0f + eps[layer]) * x[i] + agg[i];
}
__global__ void concat_kernel(const float* __restrict__ x,
                              const float* __restrict__ pe,
                              float* __restrict__ out, size_t nel) {
  const size_t i = (size_t)blockIdx.x * blockDim.x + threadIdx.x;
  if (i >= nel) return;
  const int row = (int)(i / 80), c = (int)(i % 80);
  out[i] = (c < 64) ? x[(size_t)row * 64 + c] : pe[(size_t)row * 16 + (c - 64)];
}
// f32 weights [l][K][Nc] -> f16 B-fragment swizzle [l][kt][nt][lane][e],
// K zero-padded to KT*32.
__global__ void cvtswz_kernel(const float* __restrict__ s,
                              _Float16* __restrict__ d, int K, int Nc, int KT,
                              int nlayers) {
  const size_t i = (size_t)blockIdx.x * blockDim.x + threadIdx.x;
  const int ntn = Nc / 16;
  const size_t perL = (size_t)KT * ntn * 512;
  if (i >= perL * nlayers) return;
  const int l = (int)(i / perL);
  const size_t r = i % perL;
  const int e = (int)(r & 15);
  const int lane = (int)((r >> 4) & 31);
  const int tile = (int)(r >> 9);
  const int kt = tile / ntn, nt = tile % ntn;
  const int k = kt * 32 + ((lane >> 4) << 3) + ((e < 8) ? e : (e + 8));
  const int n = nt * 16 + (lane & 15);
  d[i] = (k < K) ? (_Float16)s[((size_t)l * K + k) * Nc + n] : (_Float16)0.0f;
}
// vt[b][c][s] = qkv[b*S+s][256+c]  (transposed V stream, f16)
__global__ void vtrans_kernel(const _Float16* __restrict__ qkv,
                              _Float16* __restrict__ vt) {
  const size_t i = (size_t)blockIdx.x * blockDim.x + threadIdx.x;
  if (i >= (size_t)kB * 128 * kS) return;
  const int b = (int)(i >> 17);
  const int c = (int)((i >> 10) & 127);
  const int s = (int)(i & 1023);
  vt[i] = qkv[((size_t)b * kS + s) * 384 + 256 + c];
}
__global__ void gmean_kernel(const float* __restrict__ x, float* __restrict__ gm) {
  const int b = blockIdx.x, c = threadIdx.x;  // 128 threads
  float s = 0.0f;
  for (int i = 0; i < kS; ++i) s += x[((size_t)b * kS + i) * 128 + c];
  gm[b * 128 + c] = s * (1.0f / kS);
}
__global__ void readout_kernel(const float* __restrict__ gm,
                               const float* __restrict__ w1,
                               const float* __restrict__ b1,
                               const float* __restrict__ w2,
                               const float* __restrict__ b2,
                               float* __restrict__ out) {
  __shared__ float red[128];
  const int b = blockIdx.x, c = threadIdx.x;
  float acc = b1[c];
  for (int k = 0; k < 128; ++k) acc += gm[b * 128 + k] * w1[k * 128 + c];
  red[c] = fmaxf(acc, 0.0f) * w2[c];
  __syncthreads();
  for (int s = 64; s > 0; s >>= 1) {
    if (c < s) red[c] += red[c + s];
    __syncthreads();
  }
  if (c == 0) out[b] = red[0] + b2[0];
}

}  // namespace gps

extern "C" void kernel_launch(void* const* d_in, const int* in_sizes, int n_in,
                              void* d_out, int out_size, void* d_ws, size_t ws_size,
                              hipStream_t stream) {
  using namespace gps;
  const float* x_in  = (const float*)d_in[0];
  const int*   eidx  = (const int*)d_in[1];
  const float* eattr = (const float*)d_in[3];
  const float* lappe = (const float*)d_in[4];
  const float* nw1 = (const float*)d_in[5];  const float* nb1 = (const float*)d_in[6];
  const float* nw2 = (const float*)d_in[7];  const float* nb2 = (const float*)d_in[8];
  const float* ew1 = (const float*)d_in[9];  const float* eb1 = (const float*)d_in[10];
  const float* ew2 = (const float*)d_in[11]; const float* eb2 = (const float*)d_in[12];
  const float* eps = (const float*)d_in[13];
  const float* gw1 = (const float*)d_in[14]; const float* gb1 = (const float*)d_in[15];
  const float* gw2 = (const float*)d_in[16]; const float* gb2 = (const float*)d_in[17];
  const float* elw = (const float*)d_in[18]; const float* elb = (const float*)d_in[19];
  const float* qkvw = (const float*)d_in[20]; const float* qkvb = (const float*)d_in[21];
  const float* aow  = (const float*)d_in[22]; const float* aob  = (const float*)d_in[23];
  const float* mw1  = (const float*)d_in[24]; const float* mb1  = (const float*)d_in[25];
  const float* mw2  = (const float*)d_in[26]; const float* mb2  = (const float*)d_in[27];
  const float* l1g = (const float*)d_in[28]; const float* l1b = (const float*)d_in[29];
  const float* l2g = (const float*)d_in[30]; const float* l2b = (const float*)d_in[31];
  const float* l3g = (const float*)d_in[32]; const float* l3b = (const float*)d_in[33];
  const float* pw1 = (const float*)d_in[34]; const float* pb1 = (const float*)d_in[35];
  const float* pw2 = (const float*)d_in[36]; const float* pb2 = (const float*)d_in[37];
  const float* rw1 = (const float*)d_in[38]; const float* rb1 = (const float*)d_in[39];
  const float* rw2 = (const float*)d_in[40]; const float* rb2 = (const float*)d_in[41];

  const int* src = eidx;
  const int* dst = eidx + kE;

  // ---- workspace layout: f32 region then f16 region (~190 MB total) ----
  float* ws  = (float*)d_ws;
  float* xb  = ws;
  float* agg = xb  + (size_t)kN * kC;
  float* hlb = agg + (size_t)kN * kC;
  float* hab = hlb + (size_t)kN * kC;
  float* outb= hab + (size_t)kN * kC;
  float* catb= outb+ (size_t)kN * kC;        // [N,80]
  float* gmb = catb+ (size_t)kN * 80;        // [B,128]
  _Float16* eb16  = (_Float16*)(gmb + (size_t)kB * kC);  // [E,128]
  _Float16* big16 = eb16 + (size_t)kE * kC;  // [N,384] f16 qkv
  _Float16* vt16  = big16 + (size_t)kN * 384;            // [B,128,S] V^T
  // swizzled f16 weight arena: per [K,Nc]: KT*(Nc/16)*512 halfs
  _Float16* wa    = vt16 + (size_t)kB * 128 * kS;
  _Float16* nw1s = wa;                       // K=80 -> KT=3: 3*8*512
  _Float16* nw2s = nw1s + 3 * 8 * 512;       // 4*8*512
  _Float16* ew1s = nw2s + 4 * 8 * 512;       // K=16 -> KT=1: 1*8*512
  _Float16* ew2s = ew1s + 1 * 8 * 512;
  _Float16* gw1s = ew2s + 4 * 8 * 512;       // L x 4*8*512
  _Float16* gw2s = gw1s + 4 * (4 * 8 * 512);
  _Float16* elws = gw2s + 4 * (4 * 8 * 512);
  _Float16* qkvws = elws + 4 * (4 * 8 * 512);   // L x 4*24*512
  _Float16* aows  = qkvws + 4 * (4 * 24 * 512);
  _Float16* mw1s  = aows + 4 * (4 * 8 * 512);   // L x 4*16*512
  _Float16* mw2s  = mw1s + 4 * (4 * 16 * 512);  // L x 8*8*512
  _Float16* pw1s  = mw2s + 4 * (8 * 8 * 512);
  _Float16* pw2s  = pw1s + 4 * 8 * 512;

  const size_t NC = (size_t)kN * kC;
  auto eltg = [](size_t n) { return dim3((unsigned)((n + 255) / 256)); };
  auto swz = [&](const float* s, _Float16* d, int K, int Nc, int KT, int L) {
    const size_t n = (size_t)L * KT * (Nc / 16) * 512;
    cvtswz_kernel<<<eltg(n), 256, 0, stream>>>(s, d, K, Nc, KT, L);
  };

  // ---- weight swizzle+cvt (tiny; graph-captured each call) ----
  swz(nw1, nw1s, 80, 128, 3, 1);   swz(nw2, nw2s, 128, 128, 4, 1);
  swz(ew1, ew1s, 16, 128, 1, 1);   swz(ew2, ew2s, 128, 128, 4, 1);
  swz(gw1, gw1s, 128, 128, 4, 4);  swz(gw2, gw2s, 128, 128, 4, 4);
  swz(elw, elws, 128, 128, 4, 4);  swz(qkvw, qkvws, 128, 384, 4, 4);
  swz(aow, aows, 128, 128, 4, 4);  swz(mw1, mw1s, 128, 256, 4, 4);
  swz(mw2, mw2s, 256, 128, 8, 4);
  swz(pw1, pw1s, 128, 128, 4, 1);  swz(pw2, pw2s, 128, 128, 4, 1);

  // ---- encoders ----
  concat_kernel<<<eltg((size_t)kN * 80), 256, 0, stream>>>(x_in, lappe, catb,
                                                           (size_t)kN * 80);
  mlp2_kernel<128, ACT_RELU, false><<<kN / 64, 128, 0, stream>>>(
      catb, 80, 80, nw1s, nb1, nw2s, nb2, nullptr, xb, kN);
  mlp2_kernel<128, ACT_RELU, true><<<kE / 64, 128, 0, stream>>>(
      eattr, 16, 16, ew1s, eb1, ew2s, eb2, nullptr, eb16, kE);

  // ---- layers ----
  for (int i = 0; i < kL; ++i) {
    // local GINEConv
    zero_kernel<<<eltg(NC), 256, 0, stream>>>(agg, NC);
    gine_kernel<<<kE / 64, 128, 0, stream>>>(eb16, elws + (size_t)i * 4 * 8 * 512,
                                             elb + (size_t)i * kC, xb, src, dst,
                                             agg);
    ginpre_kernel<<<eltg(NC), 256, 0, stream>>>(xb, agg, eps, i, outb, NC);
    mlp2_kernel<128, ACT_RELU, false><<<kN / 64, 128, 0, stream>>>(
        outb, 128, 128, gw1s + (size_t)i * 4 * 8 * 512, gb1 + (size_t)i * kC,
        gw2s + (size_t)i * 4 * 8 * 512, gb2 + (size_t)i * kC, nullptr, hlb, kN);
    ln_kernel<<<kN / 8, 256, 0, stream>>>(hlb, xb, l1g + (size_t)i * kC,
                                          l1b + (size_t)i * kC, hlb, 0, kN);
    // global attention
    gemm128_kernel<ACT_NONE, true><<<dim3(kN / 64, 3), 128, 0, stream>>>(
        xb, 128, qkvws + (size_t)i * 4 * 24 * 512, 24, qkvb + (size_t)i * 384,
        nullptr, big16, 384, kN);
    vtrans_kernel<<<eltg((size_t)kB * 128 * kS), 256, 0, stream>>>(big16, vt16);
    attn_kernel<<<(kB * 4 * (kS / 16)) / 8, 256, 0, stream>>>(big16, vt16, outb);
    gemm128_kernel<ACT_NONE, false><<<dim3(kN / 64, 1), 128, 0, stream>>>(
        outb, 128, aows + (size_t)i * 4 * 8 * 512, 8, aob + (size_t)i * kC,
        nullptr, hab, 128, kN);
    ln_kernel<<<kN / 8, 256, 0, stream>>>(hab, xb, l2g + (size_t)i * kC,
                                          l2b + (size_t)i * kC, hab, 0, kN);
    // combine + FFN + norm3, outer residual
    add_kernel<<<eltg(NC), 256, 0, stream>>>(hlb, hab, outb, NC);
    mlp2_kernel<256, ACT_GELU, false><<<kN / 64, 128, 0, stream>>>(
        outb, 128, 128, mw1s + (size_t)i * 4 * 16 * 512, mb1 + (size_t)i * 256,
        mw2s + (size_t)i * 8 * 8 * 512, mb2 + (size_t)i * kC, outb, hlb, kN);
    ln_kernel<<<kN / 8, 256, 0, stream>>>(hlb, nullptr, l3g + (size_t)i * kC,
                                          l3b + (size_t)i * kC, xb, 1, kN);
  }

  // ---- post MLP + readout ----
  mlp2_kernel<128, ACT_RELU, false><<<kN / 64, 128, 0, stream>>>(
      xb, 128, 128, pw1s, pb1, pw2s, pb2, xb, outb, kN);
  gmean_kernel<<<kB, 128, 0, stream>>>(outb, gmb);
  readout_kernel<<<kB, 128, 0, stream>>>(gmb, rw1, rb1, rw2, rb2,
                                         (float*)d_out);
}
